// GPT_41936060678483
// MI455X (gfx1250) — compile-verified, matrix operand order checked by reading, not run on hardware
//
#include <hip/hip_runtime.h>
#include <math.h>
#include <stdint.h>

// ---------------------------------------------------------------------------
// Types for CDNA5 WMMA (wave32): A/B = 16 x bf16 (8 VGPRs), C/D = 8 x f32.
// ---------------------------------------------------------------------------
typedef __bf16 bf16_t;
typedef bf16_t v16bf __attribute__((ext_vector_type(16)));
typedef bf16_t v8bf  __attribute__((ext_vector_type(8)));
typedef float  v8f   __attribute__((ext_vector_type(8)));
typedef float  v4f   __attribute__((ext_vector_type(4)));
typedef unsigned int u32x4 __attribute__((ext_vector_type(4)));
typedef int          i32x8 __attribute__((ext_vector_type(8)));
typedef int          i32x4 __attribute__((ext_vector_type(4)));

#define WMMA_BF16(a, b, c) \
  __builtin_amdgcn_wmma_f32_16x16x32_bf16(false, (a), false, (b), (short)0, (c), false, false)

#if defined(__AMDGCN__) && __has_builtin(__builtin_amdgcn_tensor_load_to_lds)
#define USE_TDM 1
#else
#define USE_TDM 0
#endif

// Model dims
#define BATCH 4
#define TT    1024
#define SS    128
#define CC    1024
#define HH    16
#define NLAY  8
#define VOC   4096
#define LL    2048           // 2*T
#define MROWS 8192           // B*L
#define BT    4096           // B*T (also number of odd tokens)

// ---------------------------------------------------------------------------
// Fragment load: row index = lane&15, K packed per the CDNA5 16-bit A layout
// (lanes 0-15 hold K {hi..hi+7, 16+hi..16+hi+7} with hi = 8*(lane>=16)).
// B fragments use the identical pattern when the tile is stored with the
// B-column index as the "row" (weights are pre-transposed to [N][K]).
// ---------------------------------------------------------------------------
__device__ __forceinline__ v16bf frag_ld(const bf16_t* base, int ld) {
  const int lane = threadIdx.x & 31;
  const int row  = lane & 15;
  const int hi   = (lane >> 4) << 3;
  const bf16_t* p = base + (size_t)row * ld;
  v8bf lo = *(const v8bf*)(p + hi);
  v8bf h8 = *(const v8bf*)(p + 16 + hi);
  v16bf f;
#pragma unroll
  for (int i = 0; i < 8; ++i) { f[i] = lo[i]; f[8 + i] = h8[i]; }
  return f;
}

#if USE_TDM
// ---------------------------------------------------------------------------
// Tensor Data Mover: load a 2D tile (tile_k x tile_rows, 2-byte elements,
// row stride = stride_elems) from global to LDS at lds_off. D# group0 holds
// {count=1, lds_addr, global_addr, type=2}; group1 holds data_size=2B,
// tensor dims == tile dims (tile always in-bounds), tile dims and dim0
// stride. Issued per-wave (EXEC ignored); tracked with TENSORcnt.
// ---------------------------------------------------------------------------
__device__ __forceinline__ void tdm_tile_load(unsigned lds_off, const void* gptr,
                                              unsigned tile_k, unsigned tile_rows,
                                              unsigned stride_elems) {
  unsigned long long ga = (unsigned long long)(uintptr_t)gptr;
  u32x4 g0;
  g0[0] = 1u;                                               // count = 1
  g0[1] = lds_off;                                          // lds_addr (bytes)
  g0[2] = (unsigned)ga;                                     // global_addr[31:0]
  g0[3] = (unsigned)((ga >> 32) & 0x1FFFFFFu) | (2u << 30); // addr[56:32], type=2
  i32x8 g1;
  g1[0] = (int)(1u << 16);                                  // data_size = 2 bytes
  g1[1] = (int)((tile_k & 0xFFFFu) << 16);                  // tensor_dim0[15:0]
  g1[2] = (int)((tile_k >> 16) | ((tile_rows & 0xFFFFu) << 16)); // d0 hi, d1 lo
  g1[3] = (int)((tile_rows >> 16) | (tile_k << 16));        // d1 hi, tile_dim0
  g1[4] = (int)(tile_rows & 0xFFFFu);                       // tile_dim1 (dim2=0)
  g1[5] = (int)stride_elems;                                // tensor_dim0_stride
  g1[6] = 0;
  g1[7] = 0;
  i32x4 z4 = {0, 0, 0, 0};
#if __clang_major__ >= 23
  i32x8 z8 = {0, 0, 0, 0, 0, 0, 0, 0};
  __builtin_amdgcn_tensor_load_to_lds(g0, g1, z4, z4, z8, 0);
#else
  __builtin_amdgcn_tensor_load_to_lds(g0, g1, z4, z4, 0);
#endif
}
#endif  // USE_TDM

// ---------------------------------------------------------------------------
// fp32 -> bf16 conversion (states). n must be a multiple of 1024.
// ---------------------------------------------------------------------------
__global__ __launch_bounds__(256) void cvt_kernel(const float* __restrict__ in,
                                                  bf16_t* __restrict__ out, long n) {
  long i = ((long)blockIdx.x * 256 + threadIdx.x) * 4;
  if (i < n) {
    v4f v = *(const v4f*)&in[i];
#pragma unroll
    for (int j = 0; j < 4; ++j) out[i + j] = (bf16_t)v[j];
  }
}

// ---------------------------------------------------------------------------
// fp32 -> bf16 transpose-convert: out[n*K + k] = (bf16) in[k*N + n].
// 32x32 LDS tile, block = 32x8 threads. Run once per weight matrix.
// ---------------------------------------------------------------------------
__global__ __launch_bounds__(256) void cvt_t_kernel(const float* __restrict__ in,
                                                    bf16_t* __restrict__ out,
                                                    int K, int N) {
  __shared__ float tile[32][33];
  const int k0 = blockIdx.y * 32, n0 = blockIdx.x * 32;
  const int tx = threadIdx.x & 31, ty = threadIdx.x >> 5;
  for (int r = ty; r < 32; r += 8)
    tile[r][tx] = in[(size_t)(k0 + r) * N + n0 + tx];
  __syncthreads();
  for (int r = ty; r < 32; r += 8)
    out[(size_t)(n0 + r) * K + k0 + tx] = (bf16_t)tile[tx][r];
}

// ---------------------------------------------------------------------------
// GEMM:  D[M,N] = act(A[M,K](bf16) @ WT[N,K](bf16)^T + bias) + resid
// Block tile 128x64, 4 waves; each wave owns 32 rows x 64 cols = 2x4 WMMA
// 16x16x32 (8 wmma per k-step). Tiles are contiguous [row][32] slabs, DMA'd
// by the Tensor Data Mover with double buffering (wave 0 issues, TENSORcnt
// ordering, workgroup barrier publishes). act: 0 = none, 2 = exact GELU.
// ---------------------------------------------------------------------------
__global__ __launch_bounds__(128) void gemm_bf16_kernel(
    const bf16_t* __restrict__ A, const bf16_t* __restrict__ WT,
    const float* __restrict__ bias, const float* __restrict__ resid,
    float* __restrict__ outF, bf16_t* __restrict__ outB,
    int M, int N, int K, int act) {
  __shared__ bf16_t As[2][128 * 32];
  __shared__ bf16_t Bs[2][64 * 32];
  const int tid  = threadIdx.x;
  const int wid  = tid >> 5;
  const int lane = tid & 31;
  const int M0 = blockIdx.y * 128;
  const int N0 = blockIdx.x * 64;
  const int wm = wid * 32;

  v8f acc[2][4];
#pragma unroll
  for (int i = 0; i < 2; ++i)
#pragma unroll
    for (int j = 0; j < 4; ++j)
#pragma unroll
      for (int g = 0; g < 8; ++g) acc[i][j][g] = 0.0f;

  const int nk = K / 32;
#if USE_TDM
  if (wid == 0) {
    tdm_tile_load((unsigned)(uintptr_t)&As[0][0], A + (size_t)M0 * K, 32, 128, (unsigned)K);
    tdm_tile_load((unsigned)(uintptr_t)&Bs[0][0], WT + (size_t)N0 * K, 32, 64, (unsigned)K);
  }
#endif
  for (int i = 0; i < nk; ++i) {
    const int buf = i & 1;
#if USE_TDM
    if (wid == 0) {
      if (i + 1 < nk) {
        const int k1 = (i + 1) * 32;
        tdm_tile_load((unsigned)(uintptr_t)&As[buf ^ 1][0],
                      A + (size_t)M0 * K + k1, 32, 128, (unsigned)K);
        tdm_tile_load((unsigned)(uintptr_t)&Bs[buf ^ 1][0],
                      WT + (size_t)N0 * K + k1, 32, 64, (unsigned)K);
        __builtin_amdgcn_s_wait_tensorcnt(2);   // current tile's 2 DMAs done
      } else {
        __builtin_amdgcn_s_wait_tensorcnt(0);
      }
    }
    __syncthreads();
#else
    {
      const int k0 = i * 32;
#pragma unroll
      for (int c8 = 0; c8 < 4; ++c8) {
        int c = tid + c8 * 128;                 // 512 chunks of 8 for A
        int r = c >> 2, col = (c & 3) << 3;
        *(v8bf*)&As[buf][r * 32 + col] = *(const v8bf*)&A[(size_t)(M0 + r) * K + k0 + col];
      }
#pragma unroll
      for (int c8 = 0; c8 < 2; ++c8) {
        int c = tid + c8 * 128;                 // 256 chunks of 8 for B
        int r = c >> 2, col = (c & 3) << 3;
        *(v8bf*)&Bs[buf][r * 32 + col] = *(const v8bf*)&WT[(size_t)(N0 + r) * K + k0 + col];
      }
      __syncthreads();
    }
#endif
    v16bf a0 = frag_ld(&As[buf][(wm + 0) * 32], 32);
    v16bf a1 = frag_ld(&As[buf][(wm + 16) * 32], 32);
#pragma unroll
    for (int j = 0; j < 4; ++j) {
      v16bf b = frag_ld(&Bs[buf][(j * 16) * 32], 32);
      acc[0][j] = WMMA_BF16(a0, b, acc[0][j]);
      acc[1][j] = WMMA_BF16(a1, b, acc[1][j]);
    }
    __syncthreads();
  }

  // Epilogue. C/D layout: VGPR g, lanes 0-15 -> M=g, lanes 16-31 -> M=g+8.
  const int nloc = lane & 15;
  const int rb   = (lane >> 4) << 3;
#pragma unroll
  for (int i = 0; i < 2; ++i)
#pragma unroll
    for (int j = 0; j < 4; ++j) {
      int nn = N0 + j * 16 + nloc;
      float bv = bias ? bias[nn] : 0.0f;
#pragma unroll
      for (int g = 0; g < 8; ++g) {
        int mm = M0 + wm + i * 16 + rb + g;
        float v = acc[i][j][g] + bv;
        if (act == 2) v = 0.5f * v * (1.0f + erff(v * 0.70710678118f));
        size_t idx = (size_t)mm * N + nn;
        if (resid) v += resid[idx];
        if (outF) outF[idx] = v;
        if (outB) outB[idx] = (bf16_t)v;
      }
    }
}

// ---------------------------------------------------------------------------
// Flash attention: block = (qt, head, batch), 4 waves, each wave owns 16
// query rows of a 64-row tile. Dh=64, key blocks of 32, causal, online
// softmax. Q frags live in registers; K tile [key][d] and V transposed
// [d][key] staged in LDS; P re-laid-out through per-wave LDS into A-frags.
// ---------------------------------------------------------------------------
__global__ __launch_bounds__(128) void attn_kernel(
    const bf16_t* __restrict__ q, const bf16_t* __restrict__ k,
    const bf16_t* __restrict__ v, bf16_t* __restrict__ y, float scale) {
  __shared__ bf16_t Ks[32 * 64];
  __shared__ bf16_t Vs[64 * 40];
  __shared__ bf16_t Ps[4][16 * 32];
  const int tid = threadIdx.x, wid = tid >> 5, lane = tid & 31;
  const int qt = blockIdx.x, h = blockIdx.y, b = blockIdx.z;
  const size_t tokbase = (size_t)b * LL;
  const int hcol = h * 64;
  const int qrow0 = qt * 64 + wid * 16;

  v16bf qa0 = frag_ld(&q[(tokbase + qrow0) * CC + hcol], CC);
  v16bf qa1 = frag_ld(&q[(tokbase + qrow0) * CC + hcol + 32], CC);

  float mrun[8], lrun[8];
  v8f o[4];
#pragma unroll
  for (int g = 0; g < 8; ++g) { mrun[g] = -__builtin_inff(); lrun[g] = 0.0f; }
#pragma unroll
  for (int d = 0; d < 4; ++d)
#pragma unroll
    for (int g = 0; g < 8; ++g) o[d][g] = 0.0f;

  const int nloc = lane & 15;
  const int rb   = (lane >> 4) << 3;
  const int nkb  = 2 * qt + 2;   // key blocks covering keys 0 .. qt*64+63

  for (int kt = 0; kt < nkb; ++kt) {
    const int kbase = kt * 32;
    // Cooperative load: K tile [32 keys][64 d], V transposed [64 d][32 keys].
#pragma unroll
    for (int i = 0; i < 2; ++i) {
      int c = tid + i * 128;
      int kr = c >> 3, d8 = (c & 7) << 3;
      *(v8bf*)&Ks[kr * 64 + d8] =
          *(const v8bf*)&k[(tokbase + kbase + kr) * CC + hcol + d8];
      v8bf vv = *(const v8bf*)&v[(tokbase + kbase + kr) * CC + hcol + d8];
#pragma unroll
      for (int j = 0; j < 8; ++j) Vs[(d8 + j) * 40 + kr] = vv[j];
    }
    if (kt + 1 < nkb) {  // prefetch next key block (global_prefetch_b8)
      __builtin_prefetch(&k[(tokbase + kbase + 32 + (tid >> 2)) * CC + hcol], 0, 1);
      __builtin_prefetch(&v[(tokbase + kbase + 32 + (tid >> 2)) * CC + hcol], 0, 1);
    }
    __syncthreads();

    // S(16x32) = Q(16x64) @ K^T : two 16-key subtiles, d chained in two wmma.
    v8f s[2];
#pragma unroll
    for (int j = 0; j < 2; ++j) {
#pragma unroll
      for (int g = 0; g < 8; ++g) s[j][g] = 0.0f;
      v16bf kb0 = frag_ld(&Ks[(j * 16) * 64], 64);
      v16bf kb1 = frag_ld(&Ks[(j * 16) * 64 + 32], 64);
      s[j] = WMMA_BF16(qa0, kb0, s[j]);
      s[j] = WMMA_BF16(qa1, kb1, s[j]);
    }

    // Scale + causal mask + per-row online softmax update.
    float mnew[8], alpha[8];
#pragma unroll
    for (int g = 0; g < 8; ++g) {
      int qg = qrow0 + rb + g;
#pragma unroll
      for (int j = 0; j < 2; ++j) {
        int kg = kbase + j * 16 + nloc;
        float sv = s[j][g] * scale;
        s[j][g] = (kg <= qg) ? sv : -3.0e38f;
      }
      float rm = fmaxf(s[0][g], s[1][g]);
#pragma unroll
      for (int mk = 1; mk < 16; mk <<= 1) rm = fmaxf(rm, __shfl_xor(rm, mk, 32));
      mnew[g]  = fmaxf(mrun[g], rm);
      alpha[g] = __expf(mrun[g] - mnew[g]);
    }
#pragma unroll
    for (int g = 0; g < 8; ++g) {
      float rsum = 0.0f;
#pragma unroll
      for (int j = 0; j < 2; ++j) {
        float p = __expf(s[j][g] - mnew[g]);
        rsum += p;
        Ps[wid][(rb + g) * 32 + j * 16 + nloc] = (bf16_t)p;
      }
#pragma unroll
      for (int mk = 1; mk < 16; mk <<= 1) rsum += __shfl_xor(rsum, mk, 32);
      lrun[g] = lrun[g] * alpha[g] + rsum;
      mrun[g] = mnew[g];
    }
#pragma unroll
    for (int d = 0; d < 4; ++d)
#pragma unroll
      for (int g = 0; g < 8; ++g) o[d][g] *= alpha[g];

    // O += P(16x32) @ V(32x64)
    v16bf pa = frag_ld(&Ps[wid][0], 32);
#pragma unroll
    for (int d = 0; d < 4; ++d) {
      v16bf vb = frag_ld(&Vs[(d * 16) * 40], 40);
      o[d] = WMMA_BF16(pa, vb, o[d]);
    }
    __syncthreads();
  }

  // Normalize and write y[token][h*64+d] (heads concatenated == reshape).
#pragma unroll
  for (int g = 0; g < 8; ++g) {
    float inv = 1.0f / lrun[g];
    size_t row = tokbase + qrow0 + rb + g;
#pragma unroll
    for (int d = 0; d < 4; ++d)
      y[row * CC + hcol + d * 16 + nloc] = (bf16_t)(o[d][g] * inv);
  }
}

// ---------------------------------------------------------------------------
// LayerNorm: one block per output row, emits bf16. Input row index is
// blockIdx.x*row_mul + row_off (lnf uses mul=2, off=1 to gather odd tokens).
// ---------------------------------------------------------------------------
__global__ __launch_bounds__(256) void ln_kernel(
    const float* __restrict__ x, const float* __restrict__ g,
    const float* __restrict__ bb, bf16_t* __restrict__ out,
    int row_mul, int row_off) {
  __shared__ float sred[2][8];
  const int r_in = blockIdx.x * row_mul + row_off;
  const float* xr = x + (size_t)r_in * CC;
  float s = 0.0f, ss = 0.0f;
  for (int c = threadIdx.x; c < CC; c += 256) {
    float v = xr[c]; s += v; ss += v * v;
  }
#pragma unroll
  for (int m = 1; m < 32; m <<= 1) {
    s += __shfl_xor(s, m, 32); ss += __shfl_xor(ss, m, 32);
  }
  const int wid = threadIdx.x >> 5, lane = threadIdx.x & 31;
  if (lane == 0) { sred[0][wid] = s; sred[1][wid] = ss; }
  __syncthreads();
  s = 0.0f; ss = 0.0f;
#pragma unroll
  for (int i = 0; i < 8; ++i) { s += sred[0][i]; ss += sred[1][i]; }
  const float mean = s * (1.0f / CC);
  const float var  = ss * (1.0f / CC) - mean * mean;
  const float rstd = rsqrtf(var + 1e-5f);
  bf16_t* orow = out + (size_t)blockIdx.x * CC;
  for (int c = threadIdx.x; c < CC; c += 256)
    orow[c] = (bf16_t)((xr[c] - mean) * rstd * g[c] + bb[c]);
}

// ---------------------------------------------------------------------------
// Token assembly: x[b,2t]=tanh(sproj)+gpe+pos[2t]; x[b,2t+1]=tanh(emb)+gpe+pos.
// ---------------------------------------------------------------------------
__global__ __launch_bounds__(256) void embed_kernel(
    const float* __restrict__ sproj, const int* __restrict__ pre_actions,
    const int* __restrict__ timesteps, const float* __restrict__ act_table,
    const float* __restrict__ pos_emb, const float* __restrict__ gpos,
    float* __restrict__ x) {
  const int t = blockIdx.x, b = blockIdx.y;
  const int ts = timesteps[b * TT + t];
  const int ai = pre_actions[b * TT + t];
  const size_t row0 = ((size_t)b * LL + 2 * t) * CC;
  for (int c = threadIdx.x; c < CC; c += 256) {
    float gp = gpos[(size_t)ts * CC + c];
    float se = tanhf(sproj[((size_t)b * TT + t) * CC + c]);
    float ae = tanhf(act_table[(size_t)ai * CC + c]);
    x[row0 + c]      = se + gp + pos_emb[(size_t)(2 * t) * CC + c];
    x[row0 + CC + c] = ae + gp + pos_emb[(size_t)(2 * t + 1) * CC + c];
  }
}

// ---------------------------------------------------------------------------
// Host orchestration
// ---------------------------------------------------------------------------
extern "C" void kernel_launch(void* const* d_in, const int* in_sizes, int n_in,
                              void* d_out, int out_size, void* d_ws, size_t ws_size,
                              hipStream_t stream) {
  (void)in_sizes; (void)n_in; (void)out_size; (void)ws_size;
  const float* states    = (const float*)d_in[0];
  const int*   pre_act   = (const int*)d_in[1];
  const int*   timesteps = (const int*)d_in[2];
  const float* W_se      = (const float*)d_in[3];
  const float* b_se      = (const float*)d_in[4];
  const float* act_table = (const float*)d_in[5];
  const float* pos_emb   = (const float*)d_in[6];
  const float* gpos      = (const float*)d_in[7];
  const float* ln1_g = (const float*)d_in[8];
  const float* ln1_b = (const float*)d_in[9];
  const float* Wq = (const float*)d_in[10]; const float* bq = (const float*)d_in[11];
  const float* Wk = (const float*)d_in[12]; const float* bk = (const float*)d_in[13];
  const float* Wv = (const float*)d_in[14]; const float* bv = (const float*)d_in[15];
  const float* Wo = (const float*)d_in[16]; const float* bo = (const float*)d_in[17];
  const float* ln2_g = (const float*)d_in[18];
  const float* ln2_b = (const float*)d_in[19];
  const float* W1 = (const float*)d_in[20]; const float* b1 = (const float*)d_in[21];
  const float* W2 = (const float*)d_in[22]; const float* b2 = (const float*)d_in[23];
  const float* lnf_g = (const float*)d_in[24];
  const float* lnf_b = (const float*)d_in[25];
  const float* W_head = (const float*)d_in[26];

  char* base = (char*)d_ws;
  size_t off = 0;
  auto take = [&](size_t bytes) -> void* {
    void* p = base + off;
    off += (bytes + 255) & ~(size_t)255;
    return p;
  };

  // Transposed bf16 weights: [N][K] layout.
  bf16_t* wseT = (bf16_t*)take((size_t)CC * SS * 2);
  bf16_t* wqT  = (bf16_t*)take((size_t)NLAY * CC * CC * 2);
  bf16_t* wkT  = (bf16_t*)take((size_t)NLAY * CC * CC * 2);
  bf16_t* wvT  = (bf16_t*)take((size_t)NLAY * CC * CC * 2);
  bf16_t* woT  = (bf16_t*)take((size_t)NLAY * CC * CC * 2);
  bf16_t* w1T  = (bf16_t*)take((size_t)NLAY * 2 * CC * CC * 2);
  bf16_t* w2T  = (bf16_t*)take((size_t)NLAY * CC * 2 * CC * 2);
  bf16_t* whT  = (bf16_t*)take((size_t)VOC * CC * 2);
  bf16_t* st_bf = (bf16_t*)take((size_t)BT * SS * 2);
  float*  sproj = (float*) take((size_t)BT * CC * 4);
  float*  x     = (float*) take((size_t)MROWS * CC * 4);
  bf16_t* hbf   = (bf16_t*)take((size_t)MROWS * CC * 2);
  bf16_t* qbf   = (bf16_t*)take((size_t)MROWS * CC * 2);
  bf16_t* kbf   = (bf16_t*)take((size_t)MROWS * CC * 2);
  bf16_t* vbf   = (bf16_t*)take((size_t)MROWS * CC * 2);
  bf16_t* ybf   = (bf16_t*)take((size_t)MROWS * CC * 2);
  bf16_t* midbf = (bf16_t*)take((size_t)MROWS * 2 * CC * 2);
  bf16_t* hodd  = (bf16_t*)take((size_t)BT * CC * 2);

  auto cvtT = [&](const float* in, bf16_t* out, int K, int N) {
    cvt_t_kernel<<<dim3(N / 32, K / 32), dim3(256), 0, stream>>>(in, out, K, N);
  };
  cvtT(W_se, wseT, SS, CC);
  for (int l = 0; l < NLAY; ++l) {
    cvtT(Wq + (size_t)l * CC * CC, wqT + (size_t)l * CC * CC, CC, CC);
    cvtT(Wk + (size_t)l * CC * CC, wkT + (size_t)l * CC * CC, CC, CC);
    cvtT(Wv + (size_t)l * CC * CC, wvT + (size_t)l * CC * CC, CC, CC);
    cvtT(Wo + (size_t)l * CC * CC, woT + (size_t)l * CC * CC, CC, CC);
    cvtT(W1 + (size_t)l * CC * 2 * CC, w1T + (size_t)l * CC * 2 * CC, CC, 2 * CC);
    cvtT(W2 + (size_t)l * 2 * CC * CC, w2T + (size_t)l * 2 * CC * CC, 2 * CC, CC);
  }
  cvtT(W_head, whT, CC, VOC);
  cvt_kernel<<<dim3((unsigned)((long)BT * SS / 1024)), dim3(256), 0, stream>>>(
      states, st_bf, (long)BT * SS);

  auto gemm = [&](const bf16_t* A, const bf16_t* WT, const float* bias,
                  const float* resid, float* oF, bf16_t* oB,
                  int M, int N, int K, int act) {
    gemm_bf16_kernel<<<dim3(N / 64, M / 128), dim3(128), 0, stream>>>(
        A, WT, bias, resid, oF, oB, M, N, K, act);
  };

  // State encoder projection (tanh applied inside embed_kernel).
  gemm(st_bf, wseT, b_se, nullptr, sproj, nullptr, BT, CC, SS, 0);
  embed_kernel<<<dim3(TT, BATCH), dim3(256), 0, stream>>>(
      sproj, pre_act, timesteps, act_table, pos_emb, gpos, x);

  const float scale = 0.125f;  // 1/sqrt(64)
  for (int l = 0; l < NLAY; ++l) {
    const size_t wC  = (size_t)l * CC * CC;
    const size_t wFF = (size_t)l * CC * 2 * CC;
    ln_kernel<<<dim3(MROWS), dim3(256), 0, stream>>>(
        x, ln1_g + (size_t)l * CC, ln1_b + (size_t)l * CC, hbf, 1, 0);
    gemm(hbf, wqT + wC, bq + (size_t)l * CC, nullptr, nullptr, qbf, MROWS, CC, CC, 0);
    gemm(hbf, wkT + wC, bk + (size_t)l * CC, nullptr, nullptr, kbf, MROWS, CC, CC, 0);
    gemm(hbf, wvT + wC, bv + (size_t)l * CC, nullptr, nullptr, vbf, MROWS, CC, CC, 0);
    attn_kernel<<<dim3(LL / 64, HH, BATCH), dim3(128), 0, stream>>>(
        qbf, kbf, vbf, ybf, scale);
    gemm(ybf, woT + wC, bo + (size_t)l * CC, x, x, nullptr, MROWS, CC, CC, 0);
    ln_kernel<<<dim3(MROWS), dim3(256), 0, stream>>>(
        x, ln2_g + (size_t)l * CC, ln2_b + (size_t)l * CC, hbf, 1, 0);
    gemm(hbf, w1T + wFF, b1 + (size_t)l * 2 * CC, nullptr, nullptr, midbf,
         MROWS, 2 * CC, CC, 2 /*gelu*/);
    gemm(midbf, w2T + wFF, b2 + (size_t)l * CC, x, x, nullptr, MROWS, CC, 2 * CC, 0);
  }

  // Final LN over odd tokens only, then head GEMM straight into d_out.
  ln_kernel<<<dim3(BT), dim3(256), 0, stream>>>(x, lnf_g, lnf_b, hodd, 2, 1);
  gemm(hodd, whT, nullptr, nullptr, (float*)d_out, nullptr, BT, VOC, CC, 0);
}
